// MixtureOfExpertsLayer_50929722196044
// MI455X (gfx1250) — compile-verified, matrix operand order checked by reading, not run on hardware
//
#include <hip/hip_runtime.h>
#include <hip/hip_bf16.h>
#include <math.h>

// ---------- types ----------
typedef __attribute__((ext_vector_type(16))) __bf16 v16bf;
typedef __attribute__((ext_vector_type(8)))  float  v8f;
typedef __attribute__((ext_vector_type(8)))  unsigned int v8u;

#define DIM        1024
#define HID        4096
#define NEXP       8
#define NTOK       8192               // 4 * 2048
#define TILE_T     16                 // tokens per workgroup
#define TILES_PER_E (NTOK / TILE_T)   // 512
#define HCHUNK     128                // h columns per chunk

// fp32 -> bf16, round to nearest even
__device__ __forceinline__ unsigned short f2bf(float f) {
    unsigned u = __builtin_bit_cast(unsigned, f);
    unsigned r = u + 0x7FFFu + ((u >> 16) & 1u);
    return (unsigned short)(r >> 16);
}

// CDNA5 bf16 WMMA fragment coordinate map (per lane L, element m = 2p+e):
//   K = 16*q + 8*(L>=16) + 2*r + e,  where p = 4*q + r
// A K-pair (even K, K+1) is one dword -> swizzling is a dword permutation.

// ---------- trivial utility kernels ----------
__global__ void zero_f32_kernel(float* p, int n) {
    int i = blockIdx.x * blockDim.x + threadIdx.x;
    int stride = gridDim.x * blockDim.x;
    for (; i < n; i += stride) p[i] = 0.0f;
}

__global__ void zero_i32_kernel(int* p, int n) {
    int i = threadIdx.x;
    if (i < n) p[i] = 0;
}

__global__ void cvt_bf16_kernel(const float* __restrict__ in,
                                unsigned short* __restrict__ out, int n) {
    int i = blockIdx.x * blockDim.x + threadIdx.x;
    int stride = gridDim.x * blockDim.x;
    for (; i < n; i += stride) out[i] = f2bf(in[i]);
}

// ---------- weight swizzle: fp32 [K][N] -> bf16 fragment-order dwords ----------
// w1 layout: [e][ntile(256)][ks(32)][lane(32)][p(8)] dwords  (ks contiguous per n-tile)
__global__ void swz_w1_kernel(const float* __restrict__ w, unsigned int* __restrict__ dst) {
    const int total = NEXP * 256 * 32 * 32 * 8;          // 16,777,216 dwords
    int od = blockIdx.x * blockDim.x + threadIdx.x;
    if (od >= total) return;
    int p     = od & 7;
    int lane  = (od >> 3) & 31;
    int frag  = od >> 8;
    int ks    = frag & 31;
    int ntile = (frag >> 5) & 255;
    int e     = frag >> 13;
    int n  = ntile * 16 + (lane & 15);
    int Kl = ((p >> 2) << 4) + ((lane >> 4) << 3) + ((p & 3) << 1);
    int K  = ks * 32 + Kl;
    size_t s = ((size_t)e * DIM + K) * HID + n;
    unsigned lo = f2bf(w[s]);
    unsigned hi = f2bf(w[s + HID]);
    dst[od] = lo | (hi << 16);
}

// w2 layout: [e][kt(128)][nt(64)][lane(32)][p(8)] dwords  (nt contiguous per k-tile)
__global__ void swz_w2_kernel(const float* __restrict__ w, unsigned int* __restrict__ dst) {
    const int total = NEXP * 128 * 64 * 32 * 8;          // 16,777,216 dwords
    int od = blockIdx.x * blockDim.x + threadIdx.x;
    if (od >= total) return;
    int p    = od & 7;
    int lane = (od >> 3) & 31;
    int frag = od >> 8;
    int nt   = frag & 63;
    int kt   = (frag >> 6) & 127;
    int e    = frag >> 13;
    int n  = nt * 16 + (lane & 15);
    int Kl = ((p >> 2) << 4) + ((lane >> 4) << 3) + ((p & 3) << 1);
    int K  = kt * 32 + Kl;
    size_t s = ((size_t)e * HID + K) * DIM + n;
    unsigned lo = f2bf(w[s]);
    unsigned hi = f2bf(w[s + DIM]);
    dst[od] = lo | (hi << 16);
}

// ---------- router: one wave32 per token ----------
__global__ void router_kernel(const float* __restrict__ x,
                              const float* __restrict__ wr,
                              const float* __restrict__ br,
                              int* __restrict__ cnt,
                              int* __restrict__ etok,
                              float* __restrict__ egate) {
    const int token = blockIdx.x * 8 + (threadIdx.x >> 5);
    const int lane  = threadIdx.x & 31;
    if (token >= NTOK) return;

    const float* xp = x + (size_t)token * DIM;
    float acc[NEXP];
#pragma unroll
    for (int e = 0; e < NEXP; ++e) acc[e] = 0.0f;

    for (int i = lane; i < DIM; i += 32) {
        float xv = xp[i];
        const float* w = wr + (size_t)i * NEXP;
#pragma unroll
        for (int e = 0; e < NEXP; ++e) acc[e] += xv * w[e];
    }
#pragma unroll
    for (int e = 0; e < NEXP; ++e) {
#pragma unroll
        for (int off = 16; off > 0; off >>= 1)
            acc[e] += __shfl_xor(acc[e], off, 32);
    }

    if (lane == 0) {
        float l[NEXP], p[NEXP];
        float m = -1e30f;
#pragma unroll
        for (int e = 0; e < NEXP; ++e) { l[e] = acc[e] + br[e]; m = fmaxf(m, l[e]); }
#pragma unroll
        for (int e = 0; e < NEXP; ++e) p[e] = __expf(l[e] - m);
        int e0 = 0;
#pragma unroll
        for (int e = 1; e < NEXP; ++e) if (p[e] > p[e0]) e0 = e;
        int e1 = -1; float best = -1e30f;
#pragma unroll
        for (int e = 0; e < NEXP; ++e)
            if (e != e0 && p[e] > best) { best = p[e]; e1 = e; }
        float inv = 1.0f / (p[e0] + p[e1]);
        int pos0 = atomicAdd(&cnt[e0], 1);
        etok [e0 * NTOK + pos0] = token;
        egate[e0 * NTOK + pos0] = p[e0] * inv;
        int pos1 = atomicAdd(&cnt[e1], 1);
        etok [e1 * NTOK + pos1] = token;
        egate[e1 * NTOK + pos1] = p[e1] * inv;
    }
}

// ---------- aux loss ----------
__global__ void loss_kernel(const int* __restrict__ cnt, float* __restrict__ out) {
    if (threadIdx.x == 0) {
        float c[NEXP], mean = 0.0f;
#pragma unroll
        for (int e = 0; e < NEXP; ++e) { c[e] = (float)cnt[e]; mean += c[e]; }
        mean *= (1.0f / NEXP);
        float var = 0.0f;
#pragma unroll
        for (int e = 0; e < NEXP; ++e) { float d = c[e] - mean; var += d * d; }
        var *= (1.0f / (NEXP - 1));   // ddof = 1
        *out = (var / mean) * 0.01f;
    }
}

// ---------- fused expert MLP: 16-token tile, bf16 WMMA ----------
__global__ __launch_bounds__(256, 1)
void moe_mlp_kernel(const unsigned short* __restrict__ xbf,
                    const v8u* __restrict__ w1v,      // swizzled fragments
                    const v8u* __restrict__ w2v,      // swizzled fragments
                    const float* __restrict__ b1,
                    const float* __restrict__ b2,
                    const int* __restrict__ etok,
                    const float* __restrict__ egate,
                    const int* __restrict__ cnt,
                    float* __restrict__ out) {
    const int e    = blockIdx.x / TILES_PER_E;
    const int tile = blockIdx.x % TILES_PER_E;
    const int nrt  = cnt[e];
    const int base = tile * TILE_T;
    if (base >= nrt) return;

    const int tid   = threadIdx.x;
    const int lane  = tid & 31;
    const int wave  = tid >> 5;
    const int lrow  = lane & 15;
    const int mbase = (lane >> 4) << 3;      // C/D: lanes>=16 hold M = r+8

    // LDS: X tile fragment-swizzled [ks(32)][lane(32)][8 dwords] = 32 KB
    __shared__ unsigned int   ldsXs[32 * 32 * 8];
    // H tile fragment-swizzled [f(4)][lane(32)][8 dwords] = 4 KB
    __shared__ unsigned short ldsHs[4 * 32 * 16];
    __shared__ int   sTok[TILE_T];
    __shared__ float sGate[TILE_T];

    if (tid < TILE_T) {
        int idx = base + tid;
        if (idx < nrt) {
            sTok[tid]  = etok [e * NTOK + idx];
            sGate[tid] = egate[e * NTOK + idx];
        } else {
            sTok[tid]  = 0;        // dummy row: gate 0 -> adds 0.0
            sGate[tid] = 0.0f;
        }
    }
    __syncthreads();

    // Stage X tile into LDS with fragment swizzle (dword permutation)
    for (int idx = tid; idx < TILE_T * (DIM / 2); idx += 256) {
        int row = idx >> 9;                 // 512 dwords per row
        int col = idx & 511;                // dword = K-pair, K = 2*col
        const unsigned int* src =
            (const unsigned int*)(xbf + (size_t)sTok[row] * DIM);
        unsigned int v = src[col];
        int f  = col >> 4;                  // K >> 5
        int q  = (col >> 3) & 1;            // (K >> 4) & 1
        int hh = (col >> 2) & 1;            // (K >> 3) & 1
        int r  = col & 3;                   // (K >> 1) & 3
        int p  = (q << 2) + r;
        int Lt = row + (hh << 4);
        ldsXs[((f << 5) + Lt) * 8 + p] = v;
    }
    __syncthreads();

    const v8u* ldsXv = (const v8u*)ldsXs;
    const v8u* ldsHv = (const v8u*)ldsHs;

    v8f oacc[8];
#pragma unroll
    for (int j = 0; j < 8; ++j) oacc[j] = (v8f){0,0,0,0,0,0,0,0};

    // Precompute swizzled LDS address for H stores: this lane produces
    // column k = wave*16 + lrow of the H tile (gemm2's K index).
    const int hk  = wave * 16 + lrow;
    const int hf  = hk >> 5;
    const int hq  = (hk >> 4) & 1;
    const int hhh = (hk >> 3) & 1;
    const int hr  = (hk >> 1) & 3;
    const int hp  = (hq << 2) + hr;
    const int he  = hk & 1;
    const int hBase = ((hf << 5) + (hhh << 4)) * 16 + hp * 2 + he;  // + M*16

    for (int hc = 0; hc < HID / HCHUNK; ++hc) {
        // ---- GEMM1: H[16x16] for this wave's h-tile, K = full DIM ----
        const int nb    = hc * HCHUNK + wave * 16 + lrow;    // global h index
        const int ntile = hc * 8 + wave;                     // h tile 0..255
        size_t w1idx = (((size_t)e * 256 + ntile) * 32) * 32 + lane;
        v8f hacc = (v8f){0,0,0,0,0,0,0,0};
#pragma unroll 8
        for (int ks = 0; ks < DIM / 32; ++ks) {
            v16bf a = __builtin_bit_cast(v16bf, ldsXv[(ks << 5) + lane]);
            v16bf b = __builtin_bit_cast(v16bf, w1v[w1idx + (size_t)ks * 32]);
            hacc = __builtin_amdgcn_wmma_f32_16x16x32_bf16(
                false, a, false, b, (short)0, hacc, false, false);
        }
        // bias + exact GELU, store bf16 H to LDS at fragment-swizzled addrs
        float b1v = b1[(size_t)e * HID + nb];
#pragma unroll
        for (int r = 0; r < 8; ++r) {
            float v = hacc[r] + b1v;
            v = 0.5f * v * (1.0f + erff(v * 0.70710678118654752f));
            int M = r + mbase;
            ldsHs[hBase + M * 16] = f2bf(v);
        }
        __syncthreads();

        // ---- GEMM2: accumulate out[16 x 128-slice], K = this chunk (128) ----
#pragma unroll
        for (int ks = 0; ks < HCHUNK / 32; ++ks) {
            v16bf a = __builtin_bit_cast(v16bf, ldsHv[(ks << 5) + lane]);
            int kt = hc * 4 + ks;
            size_t w2idx = (((size_t)e * 128 + kt) * 64 + wave * 8) * 32 + lane;
#pragma unroll
            for (int j = 0; j < 8; ++j) {
                v16bf b = __builtin_bit_cast(v16bf, w2v[w2idx + (size_t)j * 32]);
                oacc[j] = __builtin_amdgcn_wmma_f32_16x16x32_bf16(
                    false, a, false, b, (short)0, oacc[j], false, false);
            }
        }
        __syncthreads();   // ldsHs reused next chunk
    }

    // ---- epilogue: out[tok] += gate * (acc + b2) ----
#pragma unroll
    for (int j = 0; j < 8; ++j) {
        int n = wave * 128 + j * 16 + lrow;
        float b2v = b2[(size_t)e * DIM + n];
#pragma unroll
        for (int r = 0; r < 8; ++r) {
            int M = r + mbase;
            float val = sGate[M] * (oacc[j][r] + b2v);
            atomicAdd(&out[(size_t)sTok[M] * DIM + n], val);
        }
    }
}

// ---------- host-side launch ----------
extern "C" void kernel_launch(void* const* d_in, const int* in_sizes, int n_in,
                              void* d_out, int out_size, void* d_ws, size_t ws_size,
                              hipStream_t stream) {
    const float* x   = (const float*)d_in[0];   // [4,2048,1024]
    const float* wr  = (const float*)d_in[1];   // [1024,8]
    const float* br  = (const float*)d_in[2];   // [8]
    const float* w1  = (const float*)d_in[3];   // [8,1024,4096]
    const float* b1  = (const float*)d_in[4];   // [8,4096]
    const float* w2  = (const float*)d_in[5];   // [8,4096,1024]
    const float* b2  = (const float*)d_in[6];   // [8,1024]
    float* out = (float*)d_out;                 // 8192*1024 outputs + 1 loss

    // workspace carve-up (~151 MB)
    char* p = (char*)d_ws;
    unsigned short* xbf = (unsigned short*)p; p += (size_t)NTOK * DIM * 2;
    unsigned int*   w1s = (unsigned int*)p;   p += (size_t)NEXP * DIM * HID * 2;
    unsigned int*   w2s = (unsigned int*)p;   p += (size_t)NEXP * HID * DIM * 2;
    int*   etok  = (int*)p;   p += (size_t)NEXP * NTOK * 4;
    float* egate = (float*)p; p += (size_t)NEXP * NTOK * 4;
    int*   cnt   = (int*)p;   p += 64;

    const int nOut = NTOK * DIM;                    // 8,388,608
    zero_f32_kernel<<<4096, 256, 0, stream>>>(out, nOut);
    zero_i32_kernel<<<1, 32, 0, stream>>>(cnt, NEXP);

    cvt_bf16_kernel<<<4096, 256, 0, stream>>>(x, xbf, NTOK * DIM);
    swz_w1_kernel<<<65536, 256, 0, stream>>>(w1, w1s);   // 16,777,216 dwords
    swz_w2_kernel<<<65536, 256, 0, stream>>>(w2, w2s);   // 16,777,216 dwords

    router_kernel<<<NTOK / 8, 256, 0, stream>>>(x, wr, br, cnt, etok, egate);
    loss_kernel<<<1, 32, 0, stream>>>(cnt, out + nOut);

    moe_mlp_kernel<<<NEXP * TILES_PER_E, 256, 0, stream>>>(
        xbf, (const v8u*)w1s, (const v8u*)w2s, b1, b2, etok, egate, cnt, out);
}
